// Decoder_40114994545040
// MI455X (gfx1250) — matrix operand order repaired and checked by reading, compiled, not run on hardware
//
#include <hip/hip_runtime.h>
#include <hip/hip_bf16.h>

// Problem constants (from reference)
#define BB   32
#define TT   63        // TOUT - 1 decode steps
#define DD   1024
#define EE   256
#define VV   32000
#define TIN  64
#define KZ   (EE + DD + DD)   // 2304: [x | attn | h] concat for fused z GEMM
#define KA   (2 * DD)         // 2048: [h | ctx]

typedef __attribute__((ext_vector_type(16))) __bf16 v16bf;
typedef __attribute__((ext_vector_type(8)))  float  v8f;

__device__ __forceinline__ __bf16 f2bf(float f) {
    unsigned int u = __float_as_uint(f);
    unsigned int r = (u + 0x7FFFu + ((u >> 16) & 1u)) >> 16;   // RNE
    union { unsigned short s; __bf16 b; } cv; cv.s = (unsigned short)r;
    return cv.b;
}

// ---------------- WMMA fragment loaders ----------------
// A: row-major [M,K] bf16. 16-bit A 16x32 layout (ISA 7.12.2):
// lanes 0-15: K 0..7 then 16..23 ; lanes 16-31: K 8..15 then 24..31 (row = lane%16)
__device__ __forceinline__ v16bf load_a_frag(const __bf16* A, int lda, int row0, int k0, int lane) {
    int m  = lane & 15;
    int kb = k0 + ((lane & 16) ? 8 : 0);
    const __bf16* p = A + (size_t)(row0 + m) * lda + kb;
    union { uint4 q[2]; v16bf v; } u;
    u.q[0] = *reinterpret_cast<const uint4*>(p);        // K kb..kb+7   (16B aligned)
    u.q[1] = *reinterpret_cast<const uint4*>(p + 16);   // K kb+16..kb+23
    return u.v;
}
// B: weights stored transposed BT[N][K] bf16, so per-lane K-run is contiguous.
// 16-bit B 32x16 layout: lanes 0-15: K 0..7 then 16..23 ; lanes 16-31: K 8..15 then 24..31 (col = lane%16)
__device__ __forceinline__ v16bf load_b_frag(const __bf16* BT, int ldb, int col0, int k0, int lane) {
    int n  = lane & 15;
    int kb = k0 + ((lane & 16) ? 8 : 0);
    const __bf16* p = BT + (size_t)(col0 + n) * ldb + kb;
    union { uint4 q[2]; v16bf v; } u;
    u.q[0] = *reinterpret_cast<const uint4*>(p);
    u.q[1] = *reinterpret_cast<const uint4*>(p + 16);
    return u.v;
}

// ---------------- Generic bf16 WMMA GEMM (paired M-tiles) ----------------
// Each wave computes TWO 16-row tiles (rows row0 and row0+16) for one N-tile,
// so every weight (B) fragment is fetched once and used by two WMMAs.
// C[M,N] (f32, optional) and/or Cbf (bf16, optional) = A[M,K] * BT[N,K]^T + bias
__global__ __launch_bounds__(128)
void gemm_wmma_bf16(const __bf16* __restrict__ A, int lda,
                    const __bf16* __restrict__ BT,
                    const float* __restrict__ bias,
                    float* __restrict__ C, long ldc,
                    __bf16* __restrict__ Cbf, int ldcbf,
                    int Mpairs, int Ntiles, int K) {
    int gtid = blockIdx.x * blockDim.x + threadIdx.x;
    int wid  = gtid >> 5;
    int lane = gtid & 31;
    if (wid >= Mpairs * Ntiles) return;           // wave-uniform: EXEC stays all-ones
    int mp = wid % Mpairs;
    int nt = wid / Mpairs;
    int row0 = mp * 32, col0 = nt * 16;

    v8f acc0 = {};
    v8f acc1 = {};
    for (int k0 = 0; k0 < K; k0 += 32) {
        v16bf bfr = load_b_frag(BT, K, col0, k0, lane);
        v16bf a0  = load_a_frag(A, lda, row0,      k0, lane);
        v16bf a1  = load_a_frag(A, lda, row0 + 16, k0, lane);
        acc0 = __builtin_amdgcn_wmma_f32_16x16x32_bf16(
                   false, a0, false, bfr, (short)0, acc0, false, false);
        acc1 = __builtin_amdgcn_wmma_f32_16x16x32_bf16(
                   false, a1, false, bfr, (short)0, acc1, false, false);
    }
    // D layout: VGPR r -> row r (lanes 0-15) / row r+8 (lanes 16-31); col = lane%16
    int n  = col0 + (lane & 15);
    int m0 = row0 + ((lane & 16) ? 8 : 0);
    float bia = bias ? bias[n] : 0.0f;
#pragma unroll
    for (int r = 0; r < 8; ++r) {
        float v = acc0[r] + bia;
        if (C)   C[(size_t)(m0 + r) * ldc + n] = v;
        if (Cbf) Cbf[(size_t)(m0 + r) * ldcbf + n] = f2bf(v);
    }
#pragma unroll
    for (int r = 0; r < 8; ++r) {
        float v = acc1[r] + bia;
        if (C)   C[(size_t)(m0 + 16 + r) * ldc + n] = v;
        if (Cbf) Cbf[(size_t)(m0 + 16 + r) * ldcbf + n] = f2bf(v);
    }
}

// ---------------- weight conversion: f32 [K,N] -> bf16 transposed [N][out_ld] at col offset ----------------
__global__ void conv_transpose_kernel(const float* __restrict__ in, int N, int K,
                                      __bf16* __restrict__ out, int out_ld, int koff) {
    long idx = (long)blockIdx.x * blockDim.x + threadIdx.x;
    if (idx >= (long)N * K) return;
    int n = (int)(idx / K);
    int k = (int)(idx % K);
    out[(size_t)n * out_ld + koff + k] = f2bf(in[(size_t)k * N + n]);
}

__global__ void conv_kernel(const float* __restrict__ in, __bf16* __restrict__ out, long n) {
    long idx = (long)blockIdx.x * blockDim.x + threadIdx.x;
    if (idx < n) out[idx] = f2bf(in[idx]);
}

// ---------------- per-call state init ----------------
// A_z layout per batch row: [0,256)=x_emb  [256,1280)=attn  [1280,2304)=h
__global__ void init_kernel(const float* __restrict__ h0, const float* __restrict__ c0,
                            float* __restrict__ c, __bf16* __restrict__ Az) {
    int idx = blockIdx.x * blockDim.x + threadIdx.x;
    if (idx >= BB * DD) return;
    int b = idx >> 10, d = idx & (DD - 1);
    c[idx] = c0[idx];
    Az[(size_t)b * KZ + EE + d]      = f2bf(0.0f);     // attn = 0
    Az[(size_t)b * KZ + EE + DD + d] = f2bf(h0[idx]);  // h = h0
}

__global__ void embed_kernel(const int* __restrict__ inputs, const float* __restrict__ emb,
                             __bf16* __restrict__ Az, int t) {
    int idx = blockIdx.x * blockDim.x + threadIdx.x;
    if (idx >= BB * EE) return;
    int b = idx >> 8, e = idx & (EE - 1);
    int tok = inputs[b * TT + t];
    Az[(size_t)b * KZ + e] = f2bf(emb[(size_t)tok * EE + e]);
}

__device__ __forceinline__ float sigm(float x) { return 1.0f / (1.0f + __expf(-x)); }

// z[32,4096] gate order i,f,g,o -> update c, h; emit h as bf16 into A_z and attn_cat
__global__ void gates_kernel(const float* __restrict__ z, float* __restrict__ c,
                             float* __restrict__ hbuf,
                             __bf16* __restrict__ Az, __bf16* __restrict__ attn_cat) {
    int idx = blockIdx.x * blockDim.x + threadIdx.x;
    if (idx >= BB * DD) return;
    int b = idx >> 10, d = idx & (DD - 1);
    const float* zb = z + (size_t)b * (4 * DD);
    float ig = sigm(zb[d]);
    float fg = sigm(zb[DD + d]);
    float gg = tanhf(zb[2 * DD + d]);
    float og = sigm(zb[3 * DD + d]);
    float cn = fg * c[idx] + ig * gg;
    float hn = og * tanhf(cn);
    c[idx]    = cn;
    hbuf[idx] = hn;
    __bf16 hb = f2bf(hn);
    Az[(size_t)b * KZ + EE + DD + d] = hb;   // h slot for next-step z GEMM
    attn_cat[(size_t)b * KA + d]     = hb;   // first half of [h|ctx]
}

// scaled Luong attention: scores over TIN=64, softmax, context; one block per batch row
__global__ __launch_bounds__(256)
void attn_kernel(const float* __restrict__ hbuf, const float* __restrict__ keys,
                 const float* __restrict__ memory, const float* __restrict__ scale_p,
                 __bf16* __restrict__ attn_cat) {
    int b = blockIdx.x;
    __shared__ float sc[TIN];
    __shared__ float al[TIN];
    int tid = threadIdx.x, lane = tid & 31, wave = tid >> 5;  // 8 waves
    const float* h = hbuf + (size_t)b * DD;
    for (int t = wave; t < TIN; t += 8) {
        const float* kr = keys + ((size_t)b * TIN + t) * DD;
        float s = 0.0f;
        for (int d = lane; d < DD; d += 32) s += h[d] * kr[d];
        for (int off = 16; off > 0; off >>= 1) s += __shfl_down(s, off, 32);
        if (lane == 0) sc[t] = s * scale_p[0];
    }
    __syncthreads();
    if (tid == 0) {
        float m = sc[0];
        for (int t = 1; t < TIN; ++t) m = fmaxf(m, sc[t]);
        float sum = 0.0f;
        for (int t = 0; t < TIN; ++t) { float e = __expf(sc[t] - m); al[t] = e; sum += e; }
        float inv = 1.0f / sum;
        for (int t = 0; t < TIN; ++t) al[t] *= inv;
    }
    __syncthreads();
    const float* mb = memory + (size_t)b * TIN * DD;
    for (int d = tid; d < DD; d += 256) {
        float s = 0.0f;
        for (int t = 0; t < TIN; ++t) s += al[t] * mb[(size_t)t * DD + d];
        attn_cat[(size_t)b * KA + DD + d] = f2bf(s);   // ctx half of [h|ctx]
    }
}

// ---------------- host side ----------------
extern "C" void kernel_launch(void* const* d_in, const int* in_sizes, int n_in,
                              void* d_out, int out_size, void* d_ws, size_t ws_size,
                              hipStream_t stream) {
    const int*   inputs = (const int*)  d_in[0];
    const float* h0     = (const float*)d_in[1];
    const float* c0     = (const float*)d_in[2];
    const float* memory = (const float*)d_in[3];
    const float* emb    = (const float*)d_in[4];
    const float* Wx     = (const float*)d_in[5];   // [1280, 4096]
    const float* Wh     = (const float*)d_in[6];   // [1024, 4096]
    const float* bvec   = (const float*)d_in[7];   // [4096]
    const float* Wm     = (const float*)d_in[8];   // [1024, 1024]
    const float* scale  = (const float*)d_in[9];   // [1]
    const float* Wa     = (const float*)d_in[10];  // [2048, 1024]
    const float* Wfc    = (const float*)d_in[11];  // [1024, 32000]
    const float* bfc    = (const float*)d_in[12];  // [32000]
    float* out = (float*)d_out;                    // [B, T, V]

    // workspace carve-up (256B aligned)
    char* ws = (char*)d_ws;
    size_t off = 0;
    auto carve = [&](size_t bytes) -> void* {
        off = (off + 255) & ~(size_t)255;
        void* p = ws + off; off += bytes; return p;
    };
    __bf16* WzT   = (__bf16*)carve((size_t)(4 * DD) * KZ * 2);   // [4096][2304]
    __bf16* WaT   = (__bf16*)carve((size_t)DD * KA * 2);         // [1024][2048]
    __bf16* WfcT  = (__bf16*)carve((size_t)VV * DD * 2);         // [32000][1024]
    __bf16* WmT   = (__bf16*)carve((size_t)DD * DD * 2);         // [1024][1024]
    __bf16* memB  = (__bf16*)carve((size_t)BB * TIN * DD * 2);   // [2048][1024]
    float*  keys  = (float*) carve((size_t)BB * TIN * DD * 4);   // [2048][1024]
    __bf16* Az    = (__bf16*)carve((size_t)BB * KZ * 2);         // [32][2304]
    __bf16* acat  = (__bf16*)carve((size_t)BB * KA * 2);         // [32][2048]
    float*  zbuf  = (float*) carve((size_t)BB * 4 * DD * 4);     // [32][4096]
    float*  cbuf  = (float*) carve((size_t)BB * DD * 4);
    float*  hbuf  = (float*) carve((size_t)BB * DD * 4);

    const int CT = 256;
    auto blocks_for = [](long n, int bs) { return (int)((n + bs - 1) / bs); };

    // ---- once-per-call weight conversion (bf16 + transpose to [N][K]) ----
    conv_transpose_kernel<<<blocks_for((long)(4*DD)*(EE+DD), CT), CT, 0, stream>>>(
        Wx, 4*DD, EE+DD, WzT, KZ, 0);
    conv_transpose_kernel<<<blocks_for((long)(4*DD)*DD, CT), CT, 0, stream>>>(
        Wh, 4*DD, DD, WzT, KZ, EE+DD);
    conv_transpose_kernel<<<blocks_for((long)DD*KA, CT), CT, 0, stream>>>(
        Wa, DD, KA, WaT, KA, 0);
    conv_transpose_kernel<<<blocks_for((long)VV*DD, CT), CT, 0, stream>>>(
        Wfc, VV, DD, WfcT, DD, 0);
    conv_transpose_kernel<<<blocks_for((long)DD*DD, CT), CT, 0, stream>>>(
        Wm, DD, DD, WmT, DD, 0);
    conv_kernel<<<blocks_for((long)BB*TIN*DD, CT), CT, 0, stream>>>(
        memory, memB, (long)BB*TIN*DD);

    auto gemm = [&](const __bf16* A, int lda, const __bf16* BT, const float* bias,
                    float* C, long ldc, __bf16* Cbf, int ldcbf,
                    int Mtiles, int Ntiles, int K) {
        int waves  = (Mtiles / 2) * Ntiles;     // paired M-tiles per wave
        int blocks = (waves + 3) / 4;           // 4 waves (128 threads) per block
        gemm_wmma_bf16<<<blocks, 128, 0, stream>>>(A, lda, BT, bias, C, ldc,
                                                   Cbf, ldcbf, Mtiles / 2, Ntiles, K);
    };

    // keys = memory @ Wm  -> [B*TIN, D] f32  (WMMA GEMM, Mtiles=128, Ntiles=64)
    gemm(memB, DD, WmT, nullptr, keys, DD, nullptr, 0, (BB*TIN)/16, DD/16, DD);

    // state init: c=c0, Az.attn=0, Az.h=h0
    init_kernel<<<blocks_for(BB*DD, CT), CT, 0, stream>>>(h0, c0, cbuf, Az);

    // ---- sequential decode ----
    for (int t = 0; t < TT; ++t) {
        embed_kernel<<<blocks_for(BB*EE, CT), CT, 0, stream>>>(inputs, emb, Az, t);
        // z = [x|attn|h] @ [Wx;Wh] + b : M=32, N=4096, K=2304
        gemm(Az, KZ, WzT, bvec, zbuf, 4*DD, nullptr, 0, BB/16, (4*DD)/16, KZ);
        gates_kernel<<<blocks_for(BB*DD, CT), CT, 0, stream>>>(zbuf, cbuf, hbuf, Az, acat);
        attn_kernel<<<BB, 256, 0, stream>>>(hbuf, keys, memory, scale, acat);
        // attn_new = [h|ctx] @ Wa -> bf16 straight into Az attn slot (stride KZ)
        gemm(acat, KA, WaT, nullptr, nullptr, 0, Az + EE, KZ, BB/16, DD/16, KA);
        // logits = attn_new @ Wfc + bfc -> out[:, t, :]  (row stride T*V)
        gemm(Az + EE, KZ, WfcT, bfc, out + (size_t)t * VV, (long)TT * VV,
             nullptr, 0, BB/16, VV/16, DD);
    }
    (void)in_sizes; (void)n_in; (void)out_size; (void)ws_size;
}
